// ManifoldDistance_19318762898039
// MI455X (gfx1250) — matrix operand order repaired and checked by reading, not compiled, not run on hardware
//
#include <hip/hip_runtime.h>
#include <hip/hip_bf16.h>
#include <math.h>

typedef __attribute__((ext_vector_type(16))) _Float16 v16h;
typedef __attribute__((ext_vector_type(8)))  float    v8f;
typedef __attribute__((ext_vector_type(4)))  int      v4i;

#define DD       64
#define LDH      68          // f16 LDS row stride: 34 dwords/row, gcd(34,64)=2 -> conflict-free K=lane loads
#define LDF      66          // f32 LDS row stride
#define NTH      128         // 4 waves of 32
#define EPS_SPD  1e-4f
#define INVC     (1.0f/2.6f) // spectral pre-scale c=2.6 -> eigvals in (0,1.7) subset (0,3)
#define LOGC     0.95551144502744f   // log(2.6)
#define POW2S    32.0f       // 2^5 square-root levels

#define F16BUF   (DD * LDH)              // 4352 elems = 8704 B
#define OFF_Q(i) ((i) * DD * LDH)        // five f16 buffers
#define STAGE_OFF OFF_Q(3)               // f32 staging aliases q3/q4 (dead at init)

#define GLOBAL_AS __attribute__((address_space(1)))
#define LOCAL_AS  __attribute__((address_space(3)))

#if defined(__has_builtin)
#if __has_builtin(__builtin_amdgcn_global_load_async_to_lds_b128) && \
    __has_builtin(__builtin_amdgcn_s_wait_asynccnt)
#define USE_ASYNC_LDS 1
#endif
#endif
#ifndef USE_ASYNC_LDS
#define USE_ASYNC_LDS 0
#endif

// ---- WMMA fragment loaders (CDNA5 16-bit layouts, wave32) -------------------

// A fragment 16x32 f16: lane l holds row M = l&15; kb = (l<16)?0:8;
// vgpr v (halves 2v,2v+1): K = ((v&4)?16:0) + kb + 2*(v&3) (+0,+1)
__device__ __forceinline__ v16h load_a_frag(const _Float16* M, int tr, int kc, int lane) {
  union { unsigned u[8]; v16h h; } r;
  const int row = tr * 16 + (lane & 15);
  const int kb  = (lane & 16) ? 8 : 0;
  const unsigned* p = (const unsigned*)(M + row * LDH + kc * 32);
  #pragma unroll
  for (int v = 0; v < 8; ++v) {
    const int k0 = ((v & 4) ? 16 : 0) + kb + 2 * (v & 3);
    r.u[v] = p[k0 >> 1];
  }
  return r.h;
}

// B fragment 32x16 f16: lane l holds N = 16*tc + (l&15); half s holds K = 32*kc + ((s&8)?16:0) + ((l<16)?0:8) + (s&7)
__device__ __forceinline__ v16h load_b_frag(const _Float16* M, int kc, int tc, int lane) {
  union { _Float16 e[16]; v16h h; } r;
  const int col = tc * 16 + (lane & 15);
  const int kb  = (lane & 16) ? 8 : 0;
  const _Float16* p = M + (kc * 32) * LDH + col;
  #pragma unroll
  for (int s = 0; s < 16; ++s) r.e[s] = p[(((s & 8) ? 16 : 0) + kb + (s & 7)) * LDH];
  return r.h;
}

// ---- 64x64 GEMM in LDS: C = scale*(A*B) + diag*I, f16 out -------------------
__device__ __forceinline__ void mm64(const _Float16* A, const _Float16* B, _Float16* C,
                                     float scale, float diag, int tid) {
  const int lane = tid & 31;
  const int tr   = tid >> 5;                 // wave id = tile row strip
  const v16h a0 = load_a_frag(A, tr, 0, lane);
  const v16h a1 = load_a_frag(A, tr, 1, lane);
  #pragma unroll
  for (int j = 0; j < 4; ++j) {
    const v16h b0 = load_b_frag(B, 0, j, lane);
    const v16h b1 = load_b_frag(B, 1, j, lane);
    v8f acc = {};
    acc = __builtin_amdgcn_wmma_f32_16x16x32_f16(false, a0, false, b0, (short)0, acc, false, false);
    acc = __builtin_amdgcn_wmma_f32_16x16x32_f16(false, a1, false, b1, (short)0, acc, false, false);
    const int rbase = tr * 16 + ((lane >> 4) << 3);   // C layout: M = vgpr + 8*(lane>=16)
    const int col   = j * 16 + (lane & 15);           //           N = lane&15
    #pragma unroll
    for (int r = 0; r < 8; ++r) {
      const int row = rbase + r;
      const float v = scale * acc[r] + ((row == col) ? diag : 0.0f);
      C[row * LDH + col] = (_Float16)v;
    }
  }
}

// ---- one matrix log: logm = 32*log((sym(G)/c + eps/c I)^(1/32)) + log(c) I --
// writeLA: store f32 logm into sLA. else: return sum (sLA - logm)^2 over this
// thread's output elements.
__device__ float process_log(const float* __restrict__ G,
                             _Float16* smemh, float* stage,
                             float* sLA, bool writeLA, int tid) {
  _Float16 *pY  = smemh + OFF_Q(0);
  _Float16 *pZ  = smemh + OFF_Q(1);
  _Float16 *pT  = smemh + OFF_Q(2);
  _Float16 *pY2 = smemh + OFF_Q(3);
  _Float16 *pZ2 = smemh + OFF_Q(4);

#if USE_ASYNC_LDS
  // DMA the raw 16KB matrix into LDS (ASYNCcnt path), then symmetrize from LDS.
  #pragma unroll
  for (int k = 0; k < 8; ++k) {
    const int off = (tid + k * NTH) * 4;   // 4 floats = 16B per lane
    __builtin_amdgcn_global_load_async_to_lds_b128(
        (GLOBAL_AS v4i*)(unsigned long long)(G + off),
        (LOCAL_AS v4i*)(unsigned int)(unsigned long long)(stage + off),
        0, 0);
  }
  __builtin_amdgcn_s_wait_asynccnt(0);
  __syncthreads();
  for (int e = tid; e < DD * DD; e += NTH) {
    const int i = e >> 6, j = e & 63;
    float v = 0.5f * (stage[(i << 6) + j] + stage[(j << 6) + i]);
    if (i == j) v += EPS_SPD;
    pY[i * LDH + j] = (_Float16)(v * INVC);
  }
#else
  // Y0 = (0.5*(G + G^T) + eps I) / c   (ensure_spd fast path, pre-scaled)
  for (int e = tid; e < DD * DD; e += NTH) {
    const int i = e >> 6, j = e & 63;
    float v = 0.5f * (G[(i << 6) + j] + G[(j << 6) + i]);
    if (i == j) v += EPS_SPD;
    pY[i * LDH + j] = (_Float16)(v * INVC);
  }
#endif
  __syncthreads();

  // 5 levels of Newton-Schulz matrix square root (inverse free)
  const int iters[5] = {9, 7, 6, 5, 4};
  #pragma unroll 1
  for (int lvl = 0; lvl < 5; ++lvl) {
    for (int e = tid; e < DD * DD; e += NTH) {            // Z = I
      const int i = e >> 6, j = e & 63;
      pZ[i * LDH + j] = (_Float16)((i == j) ? 1.0f : 0.0f);
    }
    __syncthreads();
    #pragma unroll 1
    for (int it = 0; it < iters[lvl]; ++it) {
      mm64(pZ, pY, pT, -0.5f, 1.5f, tid);                 // T = 1.5I - 0.5 Z*Y
      __syncthreads();
      mm64(pY, pT, pY2, 1.0f, 0.0f, tid);                 // Y <- Y*T
      mm64(pT, pZ, pZ2, 1.0f, 0.0f, tid);                 // Z <- T*Z
      __syncthreads();
      _Float16* t;
      t = pY; pY = pY2; pY2 = t;
      t = pZ; pZ = pZ2; pZ2 = t;
    }
  }

  // X = Y - I  (spectral radius <= ~0.10);  P = c8*X + c7*I  (Horner seed)
  for (int e = tid; e < DD * DD; e += NTH) {
    const int i = e >> 6, j = e & 63;
    const float x = (float)pY[i * LDH + j] - ((i == j) ? 1.0f : 0.0f);
    pT[i * LDH + j] = (_Float16)x;
    pZ[i * LDH + j] = (_Float16)(-0.125f * x + ((i == j) ? (1.0f / 7.0f) : 0.0f));
  }
  __syncthreads();

  // Horner: P <- c_k I + X*P  for k = 6..1   (c_k = (-1)^(k+1)/k)
  const float ck[7] = {0.0f, 1.0f, -0.5f, 1.0f / 3.0f, -0.25f, 0.2f, -1.0f / 6.0f};
  _Float16 *P = pZ, *Q = pZ2;
  #pragma unroll 1
  for (int k = 6; k >= 1; --k) {
    mm64(pT, P, Q, 1.0f, ck[k], tid);
    __syncthreads();
    _Float16* t = P; P = Q; Q = t;
  }

  // final: logm = 32*(X*P) + LOGC*I ; write f32 or accumulate diff^2
  float local = 0.0f;
  {
    const int lane = tid & 31;
    const int tr   = tid >> 5;
    const v16h a0 = load_a_frag(pT, tr, 0, lane);
    const v16h a1 = load_a_frag(pT, tr, 1, lane);
    #pragma unroll
    for (int j = 0; j < 4; ++j) {
      const v16h b0 = load_b_frag(P, 0, j, lane);
      const v16h b1 = load_b_frag(P, 1, j, lane);
      v8f acc = {};
      acc = __builtin_amdgcn_wmma_f32_16x16x32_f16(false, a0, false, b0, (short)0, acc, false, false);
      acc = __builtin_amdgcn_wmma_f32_16x16x32_f16(false, a1, false, b1, (short)0, acc, false, false);
      const int rbase = tr * 16 + ((lane >> 4) << 3);
      const int col   = j * 16 + (lane & 15);
      #pragma unroll
      for (int r = 0; r < 8; ++r) {
        const int row = rbase + r;
        const float lm = POW2S * acc[r] + ((row == col) ? LOGC : 0.0f);
        if (writeLA) {
          sLA[row * LDF + col] = lm;
        } else {
          const float d = sLA[row * LDF + col] - lm;
          local += d * d;
        }
      }
    }
  }
  __syncthreads();
  return local;
}

__global__ __launch_bounds__(NTH)
void ManifoldDistance_19318762898039_kernel(const float* __restrict__ gA,
                                            const float* __restrict__ gB,
                                            float* __restrict__ out) {
  // one flat LDS pool: 5 x f16[64*68] (43520B) + f32 logA[64*66] (16896B) + red
  __shared__ __attribute__((aligned(16))) _Float16 smemh[5 * DD * LDH];
  __shared__ __attribute__((aligned(16))) float    sLA[DD * LDF];
  __shared__ float sRed;

  const int tid = threadIdx.x;
  const int b   = blockIdx.x;
  const float* GA = gA + (size_t)b * DD * DD;
  const float* GB = gB + (size_t)b * DD * DD;

  // f32 staging area aliases the two ping-pong f16 buffers (dead at init time)
  float* stage = (float*)(smemh + STAGE_OFF);

  // hide B's HBM latency behind A's compute (lowers to global_prefetch_b8)
  __builtin_prefetch(GB + tid * 32, 0, 0);

  (void)process_log(GA, smemh, stage, sLA, true,  tid);
  const float local = process_log(GB, smemh, stage, sLA, false, tid);

  if (tid == 0) sRed = 0.0f;
  __syncthreads();
  atomicAdd(&sRed, local);          // ds_add_f32
  __syncthreads();
  if (tid == 0) out[b] = sqrtf(sRed);
}

extern "C" void kernel_launch(void* const* d_in, const int* in_sizes, int n_in,
                              void* d_out, int out_size, void* d_ws, size_t ws_size,
                              hipStream_t stream) {
  const float* A = (const float*)d_in[0];
  const float* B = (const float*)d_in[1];
  float* out = (float*)d_out;
  const int batch = in_sizes[0] / (DD * DD);   // 8192
  ManifoldDistance_19318762898039_kernel<<<batch, NTH, 0, stream>>>(A, B, out);
}
